// SelfAttnDecoder_60181081751874
// MI455X (gfx1250) — compile-verified, matrix operand order checked by reading, not run on hardware
//
#include <hip/hip_runtime.h>
#include <stdint.h>

// ---------------------------------------------------------------------------
// MI455X (gfx1250) transformer block: bf16 WMMA GEMMs + flash attention.
// wave32; V_WMMA_F32_16X16X32_BF16 for every matmul.
// GEMM tiles staged by the Tensor Data Mover (TENSORcnt); attention K tiles
// staged via global_load_async_to_lds_b128 (ASYNCcnt). Double buffered.
// ---------------------------------------------------------------------------

typedef __attribute__((ext_vector_type(16))) __bf16       v16bf;
typedef __attribute__((ext_vector_type(8)))  float        v8f;
typedef __attribute__((ext_vector_type(4)))  int          v4i;
typedef __attribute__((ext_vector_type(4)))  unsigned int v4u;
typedef __attribute__((ext_vector_type(8)))  unsigned int v8u;

union Frag16 { v16bf v; uint32_t u[8]; };

__device__ __forceinline__ unsigned short f2bf(float f) {
    union { float f; uint32_t u; } cv; cv.f = f;
    uint32_t r = cv.u + 0x7FFFu + ((cv.u >> 16) & 1u);   // RNE
    return (unsigned short)(r >> 16);
}
__device__ __forceinline__ uint32_t pack2bf(float a, float b) {
    return (uint32_t)f2bf(a) | ((uint32_t)f2bf(b) << 16);
}

// ---- CDNA5 async global->LDS (16B granule), with portable fallback --------
#if __has_builtin(__builtin_amdgcn_global_load_async_to_lds_b128)
#define HAS_ASYNC_LDS 1
typedef __attribute__((address_space(1))) v4i* gv4p;   // global int4*
typedef __attribute__((address_space(3))) v4i* lv4p;   // LDS int4*
__device__ __forceinline__ void async_copy16(const void* g, void* l) {
    __builtin_amdgcn_global_load_async_to_lds_b128((gv4p)g, (lv4p)l, 0, 0);
}
#else
#define HAS_ASYNC_LDS 0
__device__ __forceinline__ void async_copy16(const void* g, void* l) {
    *(uint4*)l = *(const uint4*)g;      // synchronous fallback
}
#endif

__device__ __forceinline__ void wait_asynccnt0() {
#if HAS_ASYNC_LDS
#if __has_builtin(__builtin_amdgcn_s_wait_asynccnt)
    __builtin_amdgcn_s_wait_asynccnt(0);
#else
    asm volatile("s_wait_asynccnt 0x0" ::: "memory");
#endif
#endif
}

// ---- CDNA5 Tensor Data Mover: 2D tile load, 2-group descriptor ------------
#if __has_builtin(__builtin_amdgcn_s_wait_tensorcnt)
#define HAS_TDM 1
__device__ __forceinline__ uint32_t lds_off(void* p) {
    return (uint32_t)(size_t)(__attribute__((address_space(3))) void*)p;
}
// Load a [rows x 32] bf16 tile (global row stride = stride_elems) into LDS,
// with +16B padding after every 64B row (LDS row stride 40 elems).
__device__ __forceinline__ void tdm_load_tile32(
    const void* gsrc, void* ldst, uint32_t rows,
    uint32_t dim0_elems, uint32_t stride_elems)
{
    uint64_t ga = (uint64_t)(size_t)gsrc;
    v4u g0;
    g0[0] = 1u;                                   // count=1, no gather
    g0[1] = lds_off(ldst);                        // lds_addr
    g0[2] = (uint32_t)ga;                         // global_addr[31:0]
    g0[3] = (uint32_t)(ga >> 32) | (2u << 30);    // global_addr[56:32] | type=2
    v8u g1;
    g1[0] = (1u << 16)                            // data_size = 2 bytes
          | (1u << 20)                            // pad_enable
          | (3u << 22)                            // pad_interval: 16 DWORDs
          | (3u << 25);                           // pad_amount:   4 DWORDs
    g1[1] = (dim0_elems & 0xFFFFu) << 16;         // tensor_dim0[15:0] @bit48
    g1[2] = (dim0_elems >> 16) | ((rows & 0xFFFFu) << 16);   // dim0 hi | dim1 lo
    g1[3] = (rows >> 16) | (32u << 16);           // dim1 hi | tile_dim0 = 32
    g1[4] = rows & 0xFFFFu;                       // tile_dim1 = rows (tile_dim2=0)
    g1[5] = stride_elems;                         // tensor_dim0_stride[31:0]
    g1[6] = 0u;                                   // stride hi | dim1_stride lo
    g1[7] = 0u;
    asm volatile("tensor_load_to_lds %0, %1" :: "s"(g0), "s"(g1) : "memory");
}
__device__ __forceinline__ void wait_tensorcnt0() {
    __builtin_amdgcn_s_wait_tensorcnt(0);
}
#else
#define HAS_TDM 0
#endif

// ---------------------------------------------------------------------------
// f32 -> bf16 conversion, 4 elements / thread
// ---------------------------------------------------------------------------
__global__ __launch_bounds__(256) void cvt_f32_bf16_kernel(
    const float* __restrict__ in, unsigned short* __restrict__ out, int n4)
{
    int i = blockIdx.x * 256 + threadIdx.x;
    if (i < n4) {
        float4 f = ((const float4*)in)[i];
        uint2 p;
        p.x = pack2bf(f.x, f.y);
        p.y = pack2bf(f.z, f.w);
        ((uint2*)out)[i] = p;
    }
}

// ---------------------------------------------------------------------------
// f32 [K,N] -> bf16 [N,K] transpose-convert (32x32 LDS tile). One-time cost;
// removes all in-loop LDS transposes from the GEMM hot path.
// ---------------------------------------------------------------------------
__global__ __launch_bounds__(256) void cvt_transpose_bf16_kernel(
    const float* __restrict__ in, unsigned short* __restrict__ out, int K, int N)
{
    __shared__ float T[32][33];
    const int tx = threadIdx.x & 31, ty = threadIdx.x >> 5;   // ty: 0..7
    const int k0 = blockIdx.y * 32,  n0 = blockIdx.x * 32;
#pragma unroll
    for (int i = 0; i < 4; ++i)
        T[ty + 8 * i][tx] = in[(size_t)(k0 + ty + 8 * i) * N + n0 + tx];
    __syncthreads();
#pragma unroll
    for (int i = 0; i < 4; ++i)
        out[(size_t)(n0 + ty + 8 * i) * K + k0 + tx] = f2bf(T[tx][ty + 8 * i]);
}

// ---------------------------------------------------------------------------
// bf16 GEMM: C[M,N] = A[M,K] * Bt[N,K]^T + bias, f32 accum.
// Block tile 128x128, BK=32, 8 waves, wave tile 64x32 (4x2 WMMA tiles).
// Tiles staged by TDM (one descriptor per operand, wave 0), double buffered;
// async-to-LDS fallback if TDM unavailable.
// ---------------------------------------------------------------------------
template<int OUT_BF16, int RELU>
__global__ __launch_bounds__(256) void gemm_bf16_kernel(
    const unsigned short* __restrict__ A, const unsigned short* __restrict__ Bt,
    const float* __restrict__ bias, void* __restrict__ Cout,
    int M, int N, int K)
{
    __shared__ unsigned short Asm[2][128][40];   // [buf][m][k], rows 16B-aligned
    __shared__ unsigned short Bts[2][128][40];   // [buf][n][k]

    const int tid   = threadIdx.x;
    const int lane  = tid & 31;
    const int wave  = tid >> 5;
    const int lr    = lane & 15;
    const int lhalf = lane >> 4;
    const int wm    = wave & 1;               // 2 waves along M
    const int wn    = wave >> 1;              // 4 waves along N
    const int m0    = blockIdx.y * 128;
    const int n0    = blockIdx.x * 128;

    v8f acc[4][2];
#pragma unroll
    for (int mi = 0; mi < 4; ++mi)
#pragma unroll
        for (int ni = 0; ni < 2; ++ni)
            acc[mi][ni] = (v8f){0.f,0.f,0.f,0.f,0.f,0.f,0.f,0.f};

    // dword index -> K offset inside a fragment (WMMA bf16 VGPR layout)
    int kk[8];
#pragma unroll
    for (int i = 0; i < 8; ++i)
        kk[i] = ((i < 4) ? 2*i : 16 + 2*(i - 4)) + lhalf * 8;

    const int row  = tid >> 1;                // 128 tile rows (fallback loader)
    const int cseg = (tid & 1) * 16;

    auto issue_tile = [&](int k0, int bufi) {
#if HAS_TDM
        if (wave == 0) {                      // one descriptor pair per block
            tdm_load_tile32(A  + (size_t)m0 * K + k0, &Asm[bufi][0][0], 128, K, K);
            tdm_load_tile32(Bt + (size_t)n0 * K + k0, &Bts[bufi][0][0], 128, K, K);
        }
#else
        const unsigned short* ga = A  + (size_t)(m0 + row) * K + k0 + cseg;
        const unsigned short* gb = Bt + (size_t)(n0 + row) * K + k0 + cseg;
        async_copy16(ga,     &Asm[bufi][row][cseg]);
        async_copy16(ga + 8, &Asm[bufi][row][cseg + 8]);
        async_copy16(gb,     &Bts[bufi][row][cseg]);
        async_copy16(gb + 8, &Bts[bufi][row][cseg + 8]);
#endif
    };
    auto wait_tiles = [&]() {
#if HAS_TDM
        wait_tensorcnt0();
#else
        wait_asynccnt0();
#endif
    };

    const int ntiles = K >> 5;
    issue_tile(0, 0);
    wait_tiles();
    __syncthreads();                          // tile 0 resident for all waves

    for (int t = 0; t < ntiles; ++t) {
        const int buf = t & 1;
        if (t + 1 < ntiles)                   // prefetch next tile into other buf
            issue_tile((t + 1) << 5, (t + 1) & 1);

        Frag16 af[4], bf[2];
#pragma unroll
        for (int mi = 0; mi < 4; ++mi) {
            int r = wm * 64 + mi * 16 + lr;
#pragma unroll
            for (int i = 0; i < 8; ++i)
                af[mi].u[i] = *(const uint32_t*)&Asm[buf][r][kk[i]];
        }
#pragma unroll
        for (int ni = 0; ni < 2; ++ni) {
            int c = wn * 32 + ni * 16 + lr;
#pragma unroll
            for (int i = 0; i < 8; ++i)
                bf[ni].u[i] = *(const uint32_t*)&Bts[buf][c][kk[i]];
        }
#pragma unroll
        for (int mi = 0; mi < 4; ++mi)
#pragma unroll
            for (int ni = 0; ni < 2; ++ni)
                acc[mi][ni] = __builtin_amdgcn_wmma_f32_16x16x32_bf16(
                    false, af[mi].v, false, bf[ni].v,
                    (short)0, acc[mi][ni], false, false);

        wait_tiles();                         // next tile landed
        __syncthreads();                      // all waves: landed + cur buf free
    }

    // epilogue: bias (+ReLU) -> f32 or bf16
#pragma unroll
    for (int mi = 0; mi < 4; ++mi) {
#pragma unroll
        for (int ni = 0; ni < 2; ++ni) {
            int colb = n0 + wn * 32 + ni * 16 + lr;
            float bv = bias ? bias[colb] : 0.f;
#pragma unroll
            for (int r = 0; r < 8; ++r) {
                int rowg = m0 + wm * 64 + mi * 16 + r + lhalf * 8;
                float v = acc[mi][ni][r] + bv;
                if (RELU) v = fmaxf(v, 0.f);
                if (OUT_BF16)
                    ((unsigned short*)Cout)[(size_t)rowg * N + colb] = f2bf(v);
                else
                    ((float*)Cout)[(size_t)rowg * N + colb] = v;
            }
        }
    }
}

// ---------------------------------------------------------------------------
// Flash attention (causal). qkv: bf16 [B,S,3*1024]; ctx out: bf16 [B,S,1024].
// 128 thr (4 waves); wave = 16-row Q tile; block = 64 q rows of one (b,h).
// K tile staged async (row-major == K^T B-operand); V staged transposed.
// ---------------------------------------------------------------------------
__global__ __launch_bounds__(128) void attn_kernel(
    const unsigned short* __restrict__ qkv, unsigned short* __restrict__ ctx)
{
    __shared__ unsigned short Ks [32][72];      // [kv][d]  (pad 8)
    __shared__ unsigned short Vts[64][40];      // [d][kv]  (transposed)
    __shared__ unsigned short Ps [4][16][40];   // per-wave P staging

    const int tid   = threadIdx.x;
    const int lane  = tid & 31;
    const int wave  = tid >> 5;
    const int lr    = lane & 15;
    const int lhalf = lane >> 4;

    const int bh = blockIdx.x >> 5;             // 32 q-tiles per (b,h)
    const int qt = blockIdx.x & 31;
    const int b  = bh >> 4, h = bh & 15;
    const int q0 = qt * 64;
    const int q_base = q0 + wave * 16;
    const float scale = 0.125f;                 // 1/sqrt(64)

    int kk[8];
#pragma unroll
    for (int i = 0; i < 8; ++i)
        kk[i] = ((i < 4) ? 2*i : 16 + 2*(i - 4)) + lhalf * 8;

    // Q A-fragments, head-dim chunks 0..31, 32..63 (per-lane dword gathers)
    Frag16 qf[2];
    {
        size_t base = ((size_t)(b * 2048 + q_base + lr)) * 3072 + h * 64;
#pragma unroll
        for (int kc = 0; kc < 2; ++kc)
#pragma unroll
            for (int i = 0; i < 8; ++i)
                qf[kc].u[i] = *(const uint32_t*)(qkv + base + kc * 32 + kk[i]);
    }

    v8f o[4];
#pragma unroll
    for (int dt = 0; dt < 4; ++dt) o[dt] = (v8f){0.f,0.f,0.f,0.f,0.f,0.f,0.f,0.f};
    float mrow[8], lrow[8];
#pragma unroll
    for (int r = 0; r < 8; ++r) { mrow[r] = -1e30f; lrow[r] = 0.f; }

    const int krow = tid >> 2, kcseg = (tid & 3) * 16;  // KV tile loaders
    const int kv_end = q0 + 64;

    for (int kv0 = 0; kv0 < kv_end; kv0 += 32) {
        size_t kvbase = ((size_t)(b * 2048 + kv0 + krow)) * 3072 + h * 64;
        const uint4* vg = (const uint4*)(qkv + kvbase + 2048 + kcseg);
        uint4 v0v = vg[0], v1v = vg[1];
        if (kv0 + 32 < kv_end)                  // prefetch next K rows
            __builtin_prefetch(qkv + kvbase + (size_t)32 * 3072 + 1024 + kcseg, 0, 1);

        __syncthreads();                        // previous tile consumed
        async_copy16(qkv + kvbase + 1024 + kcseg,     &Ks[krow][kcseg]);
        async_copy16(qkv + kvbase + 1024 + kcseg + 8, &Ks[krow][kcseg + 8]);
        {
            const unsigned short* s = (const unsigned short*)&v0v;
#pragma unroll
            for (int j = 0; j < 8; ++j) Vts[kcseg + j][krow] = s[j];
            s = (const unsigned short*)&v1v;
#pragma unroll
            for (int j = 0; j < 8; ++j) Vts[kcseg + 8 + j][krow] = s[j];
        }
        wait_asynccnt0();
        __syncthreads();                        // K/V published

        const bool active = (kv0 <= q_base + 15);   // fully-masked tiles skip

        if (active) {
            // ---- S = Q K^T : 2 n-tiles x 2 k-chunks of WMMA ----
            v8f st[2];
#pragma unroll
            for (int nt = 0; nt < 2; ++nt) {
                st[nt] = (v8f){0.f,0.f,0.f,0.f,0.f,0.f,0.f,0.f};
#pragma unroll
                for (int kc = 0; kc < 2; ++kc) {
                    Frag16 kf;
#pragma unroll
                    for (int i = 0; i < 8; ++i)
                        kf.u[i] = *(const uint32_t*)&Ks[nt * 16 + lr][kc * 32 + kk[i]];
                    st[nt] = __builtin_amdgcn_wmma_f32_16x16x32_bf16(
                        false, qf[kc].v, false, kf.v, (short)0, st[nt], false, false);
                }
            }
            // ---- online softmax (row reductions within 16-lane halves) ----
            float p0[8], p1[8];
#pragma unroll
            for (int r = 0; r < 8; ++r) {
                int rowq = q_base + r + lhalf * 8;
                float s0 = st[0][r] * scale; if (kv0 + lr      > rowq) s0 = -1e30f;
                float s1 = st[1][r] * scale; if (kv0 + 16 + lr > rowq) s1 = -1e30f;
                float mx = fmaxf(s0, s1);
#pragma unroll
                for (int msk = 1; msk < 16; msk <<= 1)
                    mx = fmaxf(mx, __shfl_xor(mx, msk, 32));
                float mnew  = fmaxf(mrow[r], mx);
                float alpha = __expf(mrow[r] - mnew);
                p0[r] = __expf(s0 - mnew);
                p1[r] = __expf(s1 - mnew);
                float rs = p0[r] + p1[r];
#pragma unroll
                for (int msk = 1; msk < 16; msk <<= 1)
                    rs += __shfl_xor(rs, msk, 32);
                lrow[r] = lrow[r] * alpha + rs;
                mrow[r] = mnew;
#pragma unroll
                for (int dt = 0; dt < 4; ++dt) o[dt][r] *= alpha;
            }
            // ---- stage P (C layout -> LDS) for A-fragment reload ----
#pragma unroll
            for (int r = 0; r < 8; ++r) {
                Ps[wave][r + lhalf * 8][lr]      = f2bf(p0[r]);
                Ps[wave][r + lhalf * 8][16 + lr] = f2bf(p1[r]);
            }
        }
        __syncthreads();                        // publish P (uniform barrier)

        if (active) {
            Frag16 pf;
#pragma unroll
            for (int i = 0; i < 8; ++i)
                pf.u[i] = *(const uint32_t*)&Ps[wave][lr][kk[i]];
#pragma unroll
            for (int dt = 0; dt < 4; ++dt) {
                Frag16 vf;
#pragma unroll
                for (int i = 0; i < 8; ++i)
                    vf.u[i] = *(const uint32_t*)&Vts[dt * 16 + lr][kk[i]];
                o[dt] = __builtin_amdgcn_wmma_f32_16x16x32_bf16(
                    false, pf.v, false, vf.v, (short)0, o[dt], false, false);
            }
        }
    }

    // normalize and write context (bf16, [B,S,H*64] layout)
#pragma unroll
    for (int r = 0; r < 8; ++r) {
        float inv = 1.0f / lrow[r];
        int row = q_base + r + lhalf * 8;
#pragma unroll
        for (int dt = 0; dt < 4; ++dt) {
            size_t idx = ((size_t)(b * 2048 + row)) * 1024 + h * 64 + dt * 16 + lr;
            ctx[idx] = f2bf(o[dt][r] * inv);
        }
    }
}

// ---------------------------------------------------------------------------
// Fused residual add + LayerNorm over D=1024. One row per block, 256 thr.
// ---------------------------------------------------------------------------
template<int WRITE_BF>
__global__ __launch_bounds__(256) void add_ln_kernel(
    const float* __restrict__ a, const float* __restrict__ b,
    const float* __restrict__ g, const float* __restrict__ be,
    float* __restrict__ y, unsigned short* __restrict__ ybf)
{
    __shared__ float ssum[8], ssq[8];
    const int row = blockIdx.x, tid = threadIdx.x;
    const float* ar = a + (size_t)row * 1024;
    const float* br = b + (size_t)row * 1024;

    float x[4], s = 0.f, sq = 0.f;
#pragma unroll
    for (int i = 0; i < 4; ++i) {
        x[i] = ar[tid + 256 * i] + br[tid + 256 * i];
        s  += x[i];
        sq += x[i] * x[i];
    }
#pragma unroll
    for (int msk = 1; msk < 32; msk <<= 1) {
        s  += __shfl_xor(s,  msk, 32);
        sq += __shfl_xor(sq, msk, 32);
    }
    if ((tid & 31) == 0) { ssum[tid >> 5] = s; ssq[tid >> 5] = sq; }
    __syncthreads();
    float tot = 0.f, totq = 0.f;
#pragma unroll
    for (int w = 0; w < 8; ++w) { tot += ssum[w]; totq += ssq[w]; }
    float mean = tot * (1.f / 1024.f);
    float var  = totq * (1.f / 1024.f) - mean * mean;
    float rstd = rsqrtf(var + 1e-5f);
#pragma unroll
    for (int i = 0; i < 4; ++i) {
        int c = tid + 256 * i;
        float yv = (x[i] - mean) * rstd * g[c] + be[c];
        y[(size_t)row * 1024 + c] = yv;
        if (WRITE_BF) ybf[(size_t)row * 1024 + c] = f2bf(yv);
    }
}

// ---------------------------------------------------------------------------
// Launch: QKV proj -> flash attn -> out proj -> LN1 -> FFN -> LN2
// ---------------------------------------------------------------------------
extern "C" void kernel_launch(void* const* d_in, const int* in_sizes, int n_in,
                              void* d_out, int out_size, void* d_ws, size_t ws_size,
                              hipStream_t stream)
{
    const float* Q     = (const float*)d_in[0];
    const float* Wqkv  = (const float*)d_in[1];
    const float* bqkv  = (const float*)d_in[2];
    const float* Wout  = (const float*)d_in[3];
    const float* bout  = (const float*)d_in[4];
    const float* g1    = (const float*)d_in[5];
    const float* be1   = (const float*)d_in[6];
    const float* W1    = (const float*)d_in[7];
    const float* b1    = (const float*)d_in[8];
    const float* W2    = (const float*)d_in[9];
    const float* b2    = (const float*)d_in[10];
    const float* g2    = (const float*)d_in[11];
    const float* be2   = (const float*)d_in[12];
    float* out = (float*)d_out;

    const int M = 4096;                 // B*S
    char* ws = (char*)d_ws;
    size_t off = 0;
    auto alloc = [&](size_t bytes) {
        void* p = ws + off;
        off = (off + bytes + 255) & ~(size_t)255;
        return p;
    };
    unsigned short* Qbf    = (unsigned short*)alloc((size_t)M * 1024 * 2);
    unsigned short* Wqkvt  = (unsigned short*)alloc((size_t)1024 * 3072 * 2);  // [3072,1024]
    unsigned short* Woutt  = (unsigned short*)alloc((size_t)1024 * 1024 * 2);  // [1024,1024]
    unsigned short* W1t    = (unsigned short*)alloc((size_t)1024 * 4096 * 2);  // [4096,1024]
    unsigned short* W2t    = (unsigned short*)alloc((size_t)4096 * 1024 * 2);  // [1024,4096]
    unsigned short* qkvbf  = (unsigned short*)alloc((size_t)M * 3072 * 2);
    unsigned short* ctxbf  = (unsigned short*)alloc((size_t)M * 1024 * 2);
    float*          attnf  = (float*)alloc((size_t)M * 1024 * 4);
    float*          xf     = (float*)alloc((size_t)M * 1024 * 4);
    unsigned short* xbf    = (unsigned short*)alloc((size_t)M * 1024 * 2);
    unsigned short* hbf    = (unsigned short*)alloc((size_t)M * 4096 * 2);
    float*          fff    = (float*)alloc((size_t)M * 1024 * 4);

    // activations: straight convert; weights: transpose-convert to [N,K] bf16
    cvt_f32_bf16_kernel<<<(M * 1024 / 4) / 256, 256, 0, stream>>>(Q, Qbf, M * 1024 / 4);
    cvt_transpose_bf16_kernel<<<dim3(3072 / 32, 1024 / 32), 256, 0, stream>>>(Wqkv, Wqkvt, 1024, 3072);
    cvt_transpose_bf16_kernel<<<dim3(1024 / 32, 1024 / 32), 256, 0, stream>>>(Wout, Woutt, 1024, 1024);
    cvt_transpose_bf16_kernel<<<dim3(4096 / 32, 1024 / 32), 256, 0, stream>>>(W1,   W1t,   1024, 4096);
    cvt_transpose_bf16_kernel<<<dim3(1024 / 32, 4096 / 32), 256, 0, stream>>>(W2,   W2t,   4096, 1024);

    // QKV projection: [4096,1024] x [1024,3072] -> bf16
    gemm_bf16_kernel<1, 0><<<dim3(3072 / 128, M / 128), 256, 0, stream>>>(
        Qbf, Wqkvt, bqkv, qkvbf, M, 3072, 1024);

    // causal flash attention: 32 (b,h) * 32 q-tiles
    attn_kernel<<<1024, 128, 0, stream>>>(qkvbf, ctxbf);

    // output projection -> f32
    gemm_bf16_kernel<0, 0><<<dim3(1024 / 128, M / 128), 256, 0, stream>>>(
        ctxbf, Woutt, bout, attnf, M, 1024, 1024);

    // LN1(Q + attn) -> xf (f32) + xbf (bf16)
    add_ln_kernel<1><<<M, 256, 0, stream>>>(Q, attnf, g1, be1, xf, xbf);

    // FFN up + ReLU -> bf16
    gemm_bf16_kernel<1, 1><<<dim3(4096 / 128, M / 128), 256, 0, stream>>>(
        xbf, W1t, b1, hbf, M, 4096, 1024);

    // FFN down -> f32
    gemm_bf16_kernel<0, 0><<<dim3(1024 / 128, M / 128), 256, 0, stream>>>(
        hbf, W2t, b2, fff, M, 1024, 4096);

    // LN2(x + ff) -> out
    add_ln_kernel<0><<<M, 256, 0, stream>>>(xf, fff, g2, be2, out, nullptr);
}
